// LSTMTest_52158082843056
// MI455X (gfx1250) — matrix operand order repaired and checked by reading, not compiled, not run on hardware
//
#include <hip/hip_runtime.h>

// ---------------------------------------------------------------------------
// Fused 3-layer LSTM (H=12, T=512, B=4096) on gfx1250.
// One wave32 owns 16 batch rows; all weights + h/c state live in VGPRs.
// Gates computed with V_WMMA_F32_16X16X32_BF16, one WMMA per gate per layer:
//   A (16x32 bf16) = [W_ih | W_hh | bias | 0-pad]  (M = gate row, K = cols)
//   B (32x16 bf16) = [y_in ; h_rec ; 1.0 ; 0-pad]  (N = batch)
//   C/D (16x16 f32) accumulator, C = inline 0 (bias rides in K row 24).
// Cell state c stays f32; only h/weights are bf16 -> error doesn't compound.
// 12 WMMAs + 20 ds_swizzle per timestep; <256 VGPRs (no s_set_vgpr_msb).
// ---------------------------------------------------------------------------

typedef __bf16 bf16;
typedef bf16  v16bf __attribute__((ext_vector_type(16)));
typedef float v8f   __attribute__((ext_vector_type(8)));

#define HID 12
#define SEQ 512
#define NBATCH 4096

// Swap values between lane n and lane n+16 (ds_swizzle SWAPX16: xor=0x10).
__device__ __forceinline__ float swapHalf(float x) {
  return __int_as_float(__builtin_amdgcn_ds_swizzle(__float_as_int(x), 0x401f));
}
__device__ __forceinline__ float fast_sigmoid(float x) {
  return __builtin_amdgcn_rcpf(1.0f + __builtin_amdgcn_exp2f(-1.44269504f * x));
}
__device__ __forceinline__ float fast_tanh(float x) {
  return 2.0f * __builtin_amdgcn_rcpf(1.0f + __builtin_amdgcn_exp2f(-2.88539008f * x)) - 1.0f;
}

__device__ __forceinline__ v8f wmma_bf16(v16bf a, v16bf b, v8f c) {
  return __builtin_amdgcn_wmma_f32_16x16x32_bf16(false, a, false, b, (short)0, c,
                                                 false, false);
}

// A-operand column for element e of v16bf: k = e + (lo?0:16).
// Layers 1/2: cols 0..11 = W_ih, 12..23 = W_hh, 24 = bias, else 0.
__device__ __forceinline__ float wval12(int k, int row, const float* wih,
                                        const float* whh, const float* bi,
                                        const float* bh) {
  if (k < 12)       return wih[row * HID + k];
  else if (k < 24)  return whh[row * HID + (k - 12)];
  else if (k == 24) return bi[row] + bh[row];
  else              return 0.0f;
}
// Layer 0: cols 0..11 = W_hh0, 12 = w_ih0 (x), 13 = bias, else 0.
__device__ __forceinline__ float wval0(int k, int row, const float* whh,
                                       const float* wih0, const float* bi,
                                       const float* bh) {
  if (k < 12)       return whh[row * HID + k];
  else if (k == 12) return wih0[row];
  else if (k == 13) return bi[row] + bh[row];
  else              return 0.0f;
}

// h tiles are in C/D f32 layout: reg r = row r (lanes<16) / row r+8 (lanes>=16).
// B-operand element e holds row k = e (lanes<16) or e+16 (lanes>=16).

// Layer 0 B: rows 0..11 = h0, 12 = x, 13 = 1 (bias), 14..31 = 0.
__device__ __forceinline__ v16bf buildB0(const float h[8], float xt, bool lo) {
  float s0 = swapHalf(h[0]), s1 = swapHalf(h[1]);
  float s2 = swapHalf(h[2]), s3 = swapHalf(h[3]);
  v16bf B;
  B[0] = (bf16)(lo ? h[0] : 0.0f);  B[1] = (bf16)(lo ? h[1] : 0.0f);
  B[2] = (bf16)(lo ? h[2] : 0.0f);  B[3] = (bf16)(lo ? h[3] : 0.0f);
  B[4] = (bf16)(lo ? h[4] : 0.0f);  B[5] = (bf16)(lo ? h[5] : 0.0f);
  B[6] = (bf16)(lo ? h[6] : 0.0f);  B[7] = (bf16)(lo ? h[7] : 0.0f);
  B[8]  = (bf16)(lo ? s0 : 0.0f);   B[9]  = (bf16)(lo ? s1 : 0.0f);
  B[10] = (bf16)(lo ? s2 : 0.0f);   B[11] = (bf16)(lo ? s3 : 0.0f);
  B[12] = (bf16)(lo ? xt : 0.0f);
  B[13] = (bf16)(lo ? 1.0f : 0.0f);
  B[14] = (bf16)0.0f;  B[15] = (bf16)0.0f;
  return B;
}

// Layers 1/2 B: rows 0..11 = y (layer input), 12..23 = h_rec, 24 = 1, else 0.
__device__ __forceinline__ v16bf buildB12(const float y[8], const float hr[8],
                                          bool lo) {
  float sy0 = swapHalf(y[0]), sy1 = swapHalf(y[1]);
  float sy2 = swapHalf(y[2]), sy3 = swapHalf(y[3]);
  float sr4 = swapHalf(hr[4]), sr5 = swapHalf(hr[5]);
  float sr6 = swapHalf(hr[6]), sr7 = swapHalf(hr[7]);
  v16bf B;
  // e: lanes<16 -> row e ; lanes>=16 -> row e+16
  B[0] = (bf16)(lo ? y[0] : sr4);   B[1] = (bf16)(lo ? y[1] : sr5);
  B[2] = (bf16)(lo ? y[2] : sr6);   B[3] = (bf16)(lo ? y[3] : sr7);
  B[4] = (bf16)(lo ? y[4] : hr[0]); B[5] = (bf16)(lo ? y[5] : hr[1]);
  B[6] = (bf16)(lo ? y[6] : hr[2]); B[7] = (bf16)(lo ? y[7] : hr[3]);
  B[8]  = (bf16)(lo ? sy0 : 1.0f);  // row 24 (hi) = bias row of 1.0
  B[9]  = (bf16)(lo ? sy1 : 0.0f);
  B[10] = (bf16)(lo ? sy2 : 0.0f);
  B[11] = (bf16)(lo ? sy3 : 0.0f);
  B[12] = (bf16)(lo ? hr[0] : 0.0f); B[13] = (bf16)(lo ? hr[1] : 0.0f);
  B[14] = (bf16)(lo ? hr[2] : 0.0f); B[15] = (bf16)(lo ? hr[3] : 0.0f);
  return B;
}

__device__ __forceinline__ void activate(v8f acc[4], float cst[8], float hout[8]) {
#pragma unroll
  for (int r = 0; r < 8; ++r) {
    float ig = fast_sigmoid(acc[0][r]);
    float fg = fast_sigmoid(acc[1][r]);
    float gg = fast_tanh(acc[2][r]);
    float og = fast_sigmoid(acc[3][r]);
    float c  = fg * cst[r] + ig * gg;
    cst[r]   = c;
    hout[r]  = og * fast_tanh(c);
  }
}

__global__ __launch_bounds__(32) void lstm3_fused(
    const float* __restrict__ x,
    const float* __restrict__ w_ih0, const float* __restrict__ w_hh0,
    const float* __restrict__ b_ih0, const float* __restrict__ b_hh0,
    const float* __restrict__ w_ih1, const float* __restrict__ w_hh1,
    const float* __restrict__ b_ih1, const float* __restrict__ b_hh1,
    const float* __restrict__ w_ih2, const float* __restrict__ w_hh2,
    const float* __restrict__ b_ih2, const float* __restrict__ b_hh2,
    const float* __restrict__ fc_w,  const float* __restrict__ fc_b,
    float* __restrict__ out)
{
  const int  lane  = threadIdx.x & 31;
  const bool lo    = lane < 16;
  const int  m     = lane & 15;         // A row / batch-within-tile
  const int  base  = blockIdx.x * 16;
  const bool rowok = m < HID;

  // ---------------- pack weights into bf16 A operands ----------------
  v16bf A[3][4];   // [layer][gate], 8 VGPRs each -> 96 VGPRs total
#pragma unroll
  for (int g = 0; g < 4; ++g) {
    const int row = g * HID + (rowok ? m : 0);
#pragma unroll
    for (int e = 0; e < 16; ++e) {
      const int klo = e;                        // + (lo?0:16) per A layout
      const int khi = klo + 16;
      float f0 = lo ? wval0(klo, row, w_hh0, w_ih0, b_ih0, b_hh0)
                    : wval0(khi, row, w_hh0, w_ih0, b_ih0, b_hh0);
      float f1 = lo ? wval12(klo, row, w_ih1, w_hh1, b_ih1, b_hh1)
                    : wval12(khi, row, w_ih1, w_hh1, b_ih1, b_hh1);
      float f2 = lo ? wval12(klo, row, w_ih2, w_hh2, b_ih2, b_hh2)
                    : wval12(khi, row, w_ih2, w_hh2, b_ih2, b_hh2);
      A[0][g][e] = (bf16)(rowok ? f0 : 0.0f);
      A[1][g][e] = (bf16)(rowok ? f1 : 0.0f);
      A[2][g][e] = (bf16)(rowok ? f2 : 0.0f);
    }
  }

  // ---------------- state ----------------
  float hv[3][8], cv[3][8];
#pragma unroll
  for (int l = 0; l < 3; ++l)
#pragma unroll
    for (int r = 0; r < 8; ++r) { hv[l][r] = 0.0f; cv[l][r] = 0.0f; }

  const float* xrow = x + (size_t)(base + m) * SEQ;
  const v8f zc = {};   // C = 0 -> inline constant operand

  // ---------------- recurrence ----------------
  for (int t = 0; t < SEQ; ++t) {
    float xt = xrow[t];
    v8f acc[4];

    // layer 0
    v16bf B0 = buildB0(hv[0], xt, lo);
#pragma unroll
    for (int g = 0; g < 4; ++g) acc[g] = wmma_bf16(A[0][g], B0, zc);
    activate(acc, cv[0], hv[0]);

    // layer 1
    v16bf B1 = buildB12(hv[0], hv[1], lo);
#pragma unroll
    for (int g = 0; g < 4; ++g) acc[g] = wmma_bf16(A[1][g], B1, zc);
    activate(acc, cv[1], hv[1]);

    // layer 2
    v16bf B2 = buildB12(hv[1], hv[2], lo);
#pragma unroll
    for (int g = 0; g < 4; ++g) acc[g] = wmma_bf16(A[2][g], B2, zc);
    activate(acc, cv[2], hv[2]);
  }

  // ---------------- epilogue: hn / cn / fc ----------------
  float* hn = out + NBATCH;                 // [3,4096,12]
  float* cn = hn + 3 * NBATCH * HID;        // [3,4096,12]
#pragma unroll
  for (int l = 0; l < 3; ++l) {
    size_t off = ((size_t)l * NBATCH + base + m) * HID + (lo ? 0 : 8);
    if (lo) {
      *(float4*)(hn + off)     = make_float4(hv[l][0], hv[l][1], hv[l][2], hv[l][3]);
      *(float4*)(hn + off + 4) = make_float4(hv[l][4], hv[l][5], hv[l][6], hv[l][7]);
      *(float4*)(cn + off)     = make_float4(cv[l][0], cv[l][1], cv[l][2], cv[l][3]);
      *(float4*)(cn + off + 4) = make_float4(cv[l][4], cv[l][5], cv[l][6], cv[l][7]);
    } else {   // rows 8..11 (12..15 are padding)
      *(float4*)(hn + off)     = make_float4(hv[l][0], hv[l][1], hv[l][2], hv[l][3]);
      *(float4*)(cn + off)     = make_float4(cv[l][0], cv[l][1], cv[l][2], cv[l][3]);
    }
  }

  // out[b] = dot(h2[b,:], fc_w) + fc_b  (half-wave partials, swap + add)
  float p = 0.0f;
#pragma unroll
  for (int r = 0; r < 8; ++r) {
    int   j = r + (lo ? 0 : 8);
    float w = (j < HID) ? fc_w[j] : 0.0f;
    p = fmaf(hv[2][r], w, p);
  }
  float tot = p + swapHalf(p);
  if (lo) out[base + m] = tot + fc_b[0];
}

extern "C" void kernel_launch(void* const* d_in, const int* in_sizes, int n_in,
                              void* d_out, int out_size, void* d_ws, size_t ws_size,
                              hipStream_t stream) {
  const float* x     = (const float*)d_in[0];
  const float* w_ih0 = (const float*)d_in[1];
  const float* w_hh0 = (const float*)d_in[2];
  const float* b_ih0 = (const float*)d_in[3];
  const float* b_hh0 = (const float*)d_in[4];
  const float* w_ih1 = (const float*)d_in[5];
  const float* w_hh1 = (const float*)d_in[6];
  const float* b_ih1 = (const float*)d_in[7];
  const float* b_hh1 = (const float*)d_in[8];
  const float* w_ih2 = (const float*)d_in[9];
  const float* w_hh2 = (const float*)d_in[10];
  const float* b_ih2 = (const float*)d_in[11];
  const float* b_hh2 = (const float*)d_in[12];
  const float* fc_w  = (const float*)d_in[13];
  const float* fc_b  = (const float*)d_in[14];
  float* out = (float*)d_out;

  dim3 grid(NBATCH / 16), block(32);
  lstm3_fused<<<grid, block, 0, stream>>>(x, w_ih0, w_hh0, b_ih0, b_hh0,
                                          w_ih1, w_hh1, b_ih1, b_hh1,
                                          w_ih2, w_hh2, b_ih2, b_hh2,
                                          fc_w, fc_b, out);
}
